// PosTransformerNet_87763361726889
// MI455X (gfx1250) — matrix operand order repaired and checked by reading, not compiled, not run on hardware
//
#include <hip/hip_runtime.h>
#include <hip/hip_bf16.h>

// ---------------------------------------------------------------------------
// PosTransformerNet (3-layer graph attention) for MI455X (gfx1250, wave32).
//   * Fused Q/K/V projection: one bf16 WMMA kernel, shared A fragments,
//     6 independent v_wmma_f32_16x16x32_bf16 per K-step per wave.
//   * Edge attention: wave-per-edge coalesced gather, segment-max via
//     order-preserving int atomicMax, fused exp + weighted scatter-add.
//   * Normalize+ReLU per node, graph pooling via atomicAdd, tiny MLP head.
// ---------------------------------------------------------------------------

typedef __attribute__((ext_vector_type(16))) __bf16 v16bf;
typedef __attribute__((ext_vector_type(8)))  float  v8f;

struct __attribute__((aligned(16))) U128 { unsigned int x, y, z, w; };

static __device__ __forceinline__ unsigned short f2bf(float f) {
    unsigned int u = __float_as_uint(f);
    unsigned int r = u + 0x7FFFu + ((u >> 16) & 1u);   // round-to-nearest-even
    return (unsigned short)(r >> 16);
}

// ------------------------------ fills --------------------------------------
__global__ void fill_f32(float* p, float v, long n) {
    long i = (long)blockIdx.x * blockDim.x + threadIdx.x;
    if (i < n) p[i] = v;
}
__global__ void fill_i32(int* p, int v, long n) {
    long i = (long)blockIdx.x * blockDim.x + threadIdx.x;
    if (i < n) p[i] = v;
}

// ----------------------- fp32 -> bf16 conversion ---------------------------
__global__ void cvt_bf16(const float* __restrict__ src, unsigned short* __restrict__ dst, long n) {
    long i = (long)blockIdx.x * blockDim.x + threadIdx.x;
    if (i < n) dst[i] = f2bf(src[i]);
}

// W is [K=256][N=256] row-major fp32; emit Wt[n][k] bf16 (transposed).
__global__ void prep_wt(const float* __restrict__ W, unsigned short* __restrict__ Wt) {
    int idx = blockIdx.x * blockDim.x + threadIdx.x;   // 65536 threads
    int n = idx & 255, k = idx >> 8;
    Wt[n * 256 + k] = f2bf(W[k * 256 + n]);
}

// --------------------------- fused Q/K/V GEMM ------------------------------
// {Q,K,V}[M,256] = A_bf16[M,256] @ W{q,k,v} + bias, fp32 out.
// Block: 16 rows, 8 waves; wave w covers cols [w*32, w*32+32) of all three
// outputs. A fragment is loaded once per K-step and fed to 6 WMMAs.
__global__ __launch_bounds__(256) void gemm_qkv(const unsigned short* __restrict__ Abf,
                                                const unsigned short* __restrict__ WtQ,
                                                const unsigned short* __restrict__ WtK,
                                                const unsigned short* __restrict__ WtV,
                                                const float* __restrict__ bq,
                                                const float* __restrict__ bk,
                                                const float* __restrict__ bv,
                                                float* __restrict__ Q,
                                                float* __restrict__ Ko,
                                                float* __restrict__ V, int M) {
    const int K = 256;
    const int lane = threadIdx.x & 31;
    const int wave = threadIdx.x >> 5;
    const int mbase = blockIdx.x * 16;
    const int nbase = wave * 32;
    const int half  = lane >> 4;      // half-wave select
    const int l16   = lane & 15;

    int arowi = mbase + l16; if (arowi > M - 1) arowi = M - 1;   // uniform EXEC
    const unsigned short* arow = Abf + (size_t)arowi * K;

    v8f aq0 = {}, aq1 = {}, ak0 = {}, ak1 = {}, av0 = {}, av1 = {};
    const size_t boff0 = (size_t)(nbase + l16) * K + half * 16;
    const size_t boff1 = (size_t)(nbase + 16 + l16) * K + half * 16;

    for (int k0 = 0; k0 < K; k0 += 32) {
        // A frag: half 0 -> K {0..15}; half 1 -> K {16..31}
        union { U128 u[2]; v16bf v; } af;
        af.u[0] = *(const U128*)(arow + k0 + half * 16);
        af.u[1] = *(const U128*)(arow + k0 + half * 16 + 8);

        union { U128 u[2]; v16bf v; } bq0, bq1, bk0, bk1, bv0, bv1;
        bq0.u[0] = *(const U128*)(WtQ + boff0 + k0); bq0.u[1] = *(const U128*)(WtQ + boff0 + k0 + 8);
        bq1.u[0] = *(const U128*)(WtQ + boff1 + k0); bq1.u[1] = *(const U128*)(WtQ + boff1 + k0 + 8);
        bk0.u[0] = *(const U128*)(WtK + boff0 + k0); bk0.u[1] = *(const U128*)(WtK + boff0 + k0 + 8);
        bk1.u[0] = *(const U128*)(WtK + boff1 + k0); bk1.u[1] = *(const U128*)(WtK + boff1 + k0 + 8);
        bv0.u[0] = *(const U128*)(WtV + boff0 + k0); bv0.u[1] = *(const U128*)(WtV + boff0 + k0 + 8);
        bv1.u[0] = *(const U128*)(WtV + boff1 + k0); bv1.u[1] = *(const U128*)(WtV + boff1 + k0 + 8);

        aq0 = __builtin_amdgcn_wmma_f32_16x16x32_bf16(false, af.v, false, bq0.v, (short)0, aq0, false, false);
        aq1 = __builtin_amdgcn_wmma_f32_16x16x32_bf16(false, af.v, false, bq1.v, (short)0, aq1, false, false);
        ak0 = __builtin_amdgcn_wmma_f32_16x16x32_bf16(false, af.v, false, bk0.v, (short)0, ak0, false, false);
        ak1 = __builtin_amdgcn_wmma_f32_16x16x32_bf16(false, af.v, false, bk1.v, (short)0, ak1, false, false);
        av0 = __builtin_amdgcn_wmma_f32_16x16x32_bf16(false, af.v, false, bv0.v, (short)0, av0, false, false);
        av1 = __builtin_amdgcn_wmma_f32_16x16x32_bf16(false, af.v, false, bv1.v, (short)0, av1, false, false);
    }
    // C/D layout: VGPR i -> row mbase+i+half*8, col = nbase + l16 (+16)
    const int c0 = nbase + l16;
    const float bQ0 = bq[c0], bQ1 = bq[c0 + 16];
    const float bK0 = bk[c0], bK1 = bk[c0 + 16];
    const float bV0 = bv[c0], bV1 = bv[c0 + 16];
    for (int i = 0; i < 8; ++i) {
        int row = mbase + i + half * 8;
        if (row < M) {
            size_t r = (size_t)row * K;
            Q [r + c0]      = aq0[i] + bQ0;
            Q [r + c0 + 16] = aq1[i] + bQ1;
            Ko[r + c0]      = ak0[i] + bK0;
            Ko[r + c0 + 16] = ak1[i] + bK1;
            V [r + c0]      = av0[i] + bV0;
            V [r + c0 + 16] = av1[i] + bV1;
        }
    }
}

// --------------------- edge pass 1: logits + segment max -------------------
// One wave per edge; lane covers 8 contiguous channels (head = lane>>3).
__global__ __launch_bounds__(256) void edge_logits(const float* __restrict__ q,
                                                   const float* __restrict__ k,
                                                   const int* __restrict__ src,
                                                   const int* __restrict__ dst,
                                                   float* __restrict__ alpha,
                                                   int* __restrict__ menc, int E) {
    const int lane = threadIdx.x & 31;
    const int e = blockIdx.x * 8 + (threadIdx.x >> 5);
    if (e >= E) return;                         // whole wave exits together
    const int s = src[e], d = dst[e];
    const float4* qp = (const float4*)(q + (size_t)d * 256) + lane * 2;
    const float4* kp = (const float4*)(k + (size_t)s * 256) + lane * 2;
    float4 q0 = qp[0], q1 = qp[1], k0 = kp[0], k1 = kp[1];
    float p = q0.x * k0.x + q0.y * k0.y + q0.z * k0.z + q0.w * k0.w
            + q1.x * k1.x + q1.y * k1.y + q1.z * k1.z + q1.w * k1.w;
    p += __shfl_xor(p, 1);
    p += __shfl_xor(p, 2);
    p += __shfl_xor(p, 4);                      // per-head sum in lanes 0,8,16,24
    if ((lane & 7) == 0) {
        int head = lane >> 3;
        float a = p * 0.125f;                   // 1/sqrt(64)
        alpha[(size_t)e * 4 + head] = a;
        int enc = __float_as_int(a);
        enc ^= (enc >> 31) & 0x7FFFFFFF;        // order-preserving int encoding
        atomicMax(menc + (size_t)d * 4 + head, enc);
    }
}

// ------------- edge pass 2: exp weight, z accum, weighted scatter ----------
__global__ __launch_bounds__(256) void edge_accum(const float* __restrict__ v,
                                                  const float* __restrict__ alpha,
                                                  const int* __restrict__ menc,
                                                  const int* __restrict__ src,
                                                  const int* __restrict__ dst,
                                                  float* __restrict__ z,
                                                  float* __restrict__ out, int E) {
    const int lane = threadIdx.x & 31;
    const int e = blockIdx.x * 8 + (threadIdx.x >> 5);
    if (e >= E) return;
    const int s = src[e], d = dst[e];
    const int head = lane >> 3;
    int me = menc[(size_t)d * 4 + head];
    me ^= (me >> 31) & 0x7FFFFFFF;              // decode
    float w = __expf(alpha[(size_t)e * 4 + head] - __int_as_float(me));
    if ((lane & 7) == 0) atomicAdd(z + (size_t)d * 4 + head, w);
    const float4* vp = (const float4*)(v + (size_t)s * 256) + lane * 2;
    float4 v0 = vp[0], v1 = vp[1];
    float* op = out + (size_t)d * 256 + lane * 8;
    atomicAdd(op + 0, w * v0.x); atomicAdd(op + 1, w * v0.y);
    atomicAdd(op + 2, w * v0.z); atomicAdd(op + 3, w * v0.w);
    atomicAdd(op + 4, w * v1.x); atomicAdd(op + 5, w * v1.y);
    atomicAdd(op + 6, w * v1.z); atomicAdd(op + 7, w * v1.w);
}

// ----------------------- normalize + ReLU per node -------------------------
__global__ void norm_relu(const float* __restrict__ out, const float* __restrict__ z,
                          float* __restrict__ h, long total) {
    long i = (long)blockIdx.x * blockDim.x + threadIdx.x;
    if (i >= total) return;
    long n = i >> 8;
    int head = (int)((i & 255) >> 6);
    float zz = z[n * 4 + head] + 1e-16f;
    float val = out[i] / zz;
    h[i] = val > 0.f ? val : 0.f;
}

// -------------------------- graph add-pool ---------------------------------
__global__ void pool_graph(const float* __restrict__ h, const int* __restrict__ batch,
                           float* __restrict__ g, long total) {
    long i = (long)blockIdx.x * blockDim.x + threadIdx.x;
    if (i >= total) return;
    long n = i >> 8;
    int c = (int)(i & 255);
    atomicAdd(g + (size_t)batch[n] * 256 + c, h[i]);
}

// ------------------------------ MLP head -----------------------------------
__global__ void mlp1(const float* __restrict__ g, const float* __restrict__ W1,
                     const float* __restrict__ b1, float* __restrict__ hid) {
    int t = blockIdx.x * blockDim.x + threadIdx.x;   // 64*64
    if (t >= 64 * 64) return;
    int gi = t >> 6, j = t & 63;
    float acc = b1[j];
    const float* gr = g + (size_t)gi * 256;
    for (int c = 0; c < 256; ++c) acc += gr[c] * W1[c * 64 + j];
    hid[t] = acc > 0.f ? acc : 0.f;
}
__global__ void mlp2(const float* __restrict__ hid, const float* __restrict__ W2,
                     const float* __restrict__ b2, float* __restrict__ y) {
    int t = blockIdx.x * blockDim.x + threadIdx.x;   // 64*16
    if (t >= 64 * 16) return;
    int gi = t >> 4, o = t & 15;
    float acc = b2[o];
    const float* hr = hid + (size_t)gi * 64;
    for (int j = 0; j < 64; ++j) acc += hr[j] * W2[j * 16 + o];
    y[t] = acc;
}

// ---------------------------------------------------------------------------
extern "C" void kernel_launch(void* const* d_in, const int* in_sizes, int n_in,
                              void* d_out, int out_size, void* d_ws, size_t ws_size,
                              hipStream_t stream) {
    const int C = 256;
    const long N = in_sizes[0] / C;            // 50000
    const long E = in_sizes[1] / 2;            // 800000

    const float* x   = (const float*)d_in[0];
    const int*   ei  = (const int*)d_in[1];
    const int*   src = ei;
    const int*   dst = ei + E;
    const int*   bat = (const int*)d_in[2];
    const float* Wq  = (const float*)d_in[3];
    const float* bq  = (const float*)d_in[4];
    const float* Wk  = (const float*)d_in[5];
    const float* bk  = (const float*)d_in[6];
    const float* Wv  = (const float*)d_in[7];
    const float* bv  = (const float*)d_in[8];
    const float* W1  = (const float*)d_in[9];
    const float* b1  = (const float*)d_in[10];
    const float* W2  = (const float*)d_in[11];
    const float* b2  = (const float*)d_in[12];
    float* y = (float*)d_out;

    // ---- workspace carve ----
    char* w = (char*)d_ws;
    auto alloc = [&](size_t bytes) { void* p = w; w += (bytes + 255) & ~(size_t)255; return p; };
    float*          h_f32  = (float*)         alloc((size_t)N * C * 4);
    unsigned short* h_bf16 = (unsigned short*)alloc((size_t)N * C * 2);
    float*          qbuf   = (float*)         alloc((size_t)N * C * 4);
    float*          kbuf   = (float*)         alloc((size_t)N * C * 4);
    float*          vbuf   = (float*)         alloc((size_t)N * C * 4);
    float*          outbuf = (float*)         alloc((size_t)N * C * 4);
    unsigned short* wt_q   = (unsigned short*)alloc((size_t)C * C * 2);
    unsigned short* wt_k   = (unsigned short*)alloc((size_t)C * C * 2);
    unsigned short* wt_v   = (unsigned short*)alloc((size_t)C * C * 2);
    float*          albuf  = (float*)         alloc((size_t)E * 4 * 4);
    int*            menc   = (int*)           alloc((size_t)N * 4 * 4);
    float*          zbuf   = (float*)         alloc((size_t)N * 4 * 4);
    float*          gbuf   = (float*)         alloc((size_t)64 * C * 4);
    float*          hidbuf = (float*)         alloc((size_t)64 * 64 * 4);
    (void)ws_size; (void)n_in; (void)out_size;

    const long NC = N * C;
    const int TB = 256;
    const int gNC = (int)((NC + TB - 1) / TB);
    const int gEdge = (int)((E + 7) / 8);
    const int gGemm = (int)((N + 15) / 16);

    for (int l = 0; l < 3; ++l) {
        const float* hsrc = (l == 0) ? x : h_f32;
        cvt_bf16<<<gNC, TB, 0, stream>>>(hsrc, h_bf16, NC);
        prep_wt<<<C * C / TB, TB, 0, stream>>>(Wq + (size_t)l * C * C, wt_q);
        prep_wt<<<C * C / TB, TB, 0, stream>>>(Wk + (size_t)l * C * C, wt_k);
        prep_wt<<<C * C / TB, TB, 0, stream>>>(Wv + (size_t)l * C * C, wt_v);

        gemm_qkv<<<gGemm, 256, 0, stream>>>(h_bf16, wt_q, wt_k, wt_v,
                                            bq + l * C, bk + l * C, bv + l * C,
                                            qbuf, kbuf, vbuf, (int)N);

        fill_i32<<<(int)((N * 4 + TB - 1) / TB), TB, 0, stream>>>(menc, (int)0x80000000, N * 4);
        fill_f32<<<(int)((N * 4 + TB - 1) / TB), TB, 0, stream>>>(zbuf, 0.f, N * 4);
        fill_f32<<<gNC, TB, 0, stream>>>(outbuf, 0.f, NC);

        edge_logits<<<gEdge, 256, 0, stream>>>(qbuf, kbuf, src, dst, albuf, menc, (int)E);
        edge_accum<<<gEdge, 256, 0, stream>>>(vbuf, albuf, menc, src, dst, zbuf, outbuf, (int)E);

        norm_relu<<<gNC, TB, 0, stream>>>(outbuf, zbuf, h_f32, NC);
    }

    fill_f32<<<(64 * C + TB - 1) / TB, TB, 0, stream>>>(gbuf, 0.f, 64 * C);
    pool_graph<<<gNC, TB, 0, stream>>>(h_f32, bat, gbuf, NC);
    mlp1<<<(64 * 64 + TB - 1) / TB, TB, 0, stream>>>(gbuf, W1, b1, hidbuf);
    mlp2<<<(64 * 16 + TB - 1) / TB, TB, 0, stream>>>(hidbuf, W2, b2, y);
}
